// ExpertRouter_17927193493781
// MI455X (gfx1250) — matrix operand order repaired and checked by reading, not compiled, not run on hardware
//
#include <hip/hip_runtime.h>
#include <hip/hip_bf16.h>

#define T_TOKENS     16384            // 4 * 4096
#define HID          2048
#define NEXP         64
#define TOKS         128              // tokens per block (8 wave32, 16 tokens/wave)
#define KCHUNK       32               // K elements per TDM tile / WMMA step

#define GATE_PITCH   2056             // ushorts: 2048 + 8 pad -> lane stride 4112B
#define GATE_BYTES   (NEXP * GATE_PITCH * 2)            // 263168
#define HID_PITCH_DW 36               // 32 data + 4 pad dwords per tile row
#define HID_TILE_B   (TOKS * HID_PITCH_DW * 4)          // 18432
#define HID_BUFS_B   (2 * HID_TILE_B)                   // 36864
#define LOG_PITCH    65
#define SMEM_BYTES   (GATE_BYTES + HID_BUFS_B + 512)    // 300544 < 320KB

typedef __attribute__((ext_vector_type(16))) __bf16 v16bf;
typedef __attribute__((ext_vector_type(8)))  float  v8f;
typedef unsigned int __attribute__((ext_vector_type(4))) v4u;
typedef int          __attribute__((ext_vector_type(8))) v8i;
typedef int          __attribute__((ext_vector_type(4))) v4i;

union FragB16 {
    v16bf v;
    unsigned short u[16];
    uint4 q[2];
};

__device__ __forceinline__ unsigned short f32_to_bf16_rne(float f) {
    union { float f; unsigned int u; } x;
    x.f = f;
    unsigned int r = x.u + 0x7FFFu + ((x.u >> 16) & 1u);
    return (unsigned short)(r >> 16);
}

// ---------------------------------------------------------------------------
// TDM: DMA one 128-row x 32-f32 tile (row stride HID) into LDS with 4-dword
// row padding (D# pad_interval=32dw, pad_amount=4dw -> 36-dword LDS pitch).
// Leader wave only; completion via TENSORcnt.
// ---------------------------------------------------------------------------
#if __has_builtin(__builtin_amdgcn_tensor_load_to_lds)
#define HAVE_TDM 1
#else
#define HAVE_TDM 0
#endif

__device__ __forceinline__ void tdm_load_tile(const float* __restrict__ hs,
                                              int tok_base, int k0,
                                              unsigned lds_addr,
                                              float* __restrict__ lds_ptr) {
#if HAVE_TDM
    (void)lds_ptr;
    unsigned long long ga = (unsigned long long)(uintptr_t)hs +
        ((unsigned long long)tok_base * HID + (unsigned)k0) * 4ull;
    v4u g0;
    g0[0] = 1u;                                   // count=1, user descriptor
    g0[1] = lds_addr;                             // LDS byte address
    g0[2] = (unsigned)ga;                         // global_addr[31:0]
    g0[3] = (unsigned)((ga >> 32) & 0x1FFFFFFull) // global_addr[56:32]
            | (2u << 30);                         // type = 2 ("image")
    v8i g1;
    g1[0] = (int)((2u << 16)        // data_size = 4B
                | (1u << 20)        // pad_enable
                | (4u << 22)        // pad_interval: 2<<4 = 32 dwords
                | (3u << 25));      // pad_amount: 3+1 = 4 dwords
    g1[1] = (int)((unsigned)HID << 16);        // tensor_dim0 = 2048
    g1[2] = (int)((unsigned)T_TOKENS << 16);   // tensor_dim1 = 16384
    g1[3] = (int)((unsigned)KCHUNK << 16);     // tile_dim0 = 32
    g1[4] = TOKS;                              // tile_dim1 = 128, tile_dim2 = 0
    g1[5] = HID;                               // tensor_dim0_stride = 2048
    g1[6] = 0;
    g1[7] = 0;
    v4i z4 = {0, 0, 0, 0};
    v8i z8 = {0, 0, 0, 0, 0, 0, 0, 0};
    __builtin_amdgcn_tensor_load_to_lds(g0, g1, z4, z4, z8, 0);
#else
    // Fallback: leader wave copies the tile manually through VGPRs.
    (void)lds_addr;
    const int lane = (int)(threadIdx.x & 31);
    for (int j = lane; j < TOKS * (KCHUNK / 4); j += 32) {
        int row = j >> 3;                       // 8 float4 per row
        int c4  = j & 7;
        float4 v = ((const float4*)(hs + (size_t)(tok_base + row) * HID + k0))[c4];
        ((float4*)((char*)lds_ptr + row * HID_PITCH_DW * 4))[c4] = v;
    }
#endif
}

__device__ __forceinline__ void tdm_wait() {
#if HAVE_TDM
    __builtin_amdgcn_s_wait_tensorcnt(0);
#endif
}

// ---------------------------------------------------------------------------
// workspace zeroing (graph-replay safe)
// ---------------------------------------------------------------------------
__global__ void router_zero_ws(float* __restrict__ ws) {
    ws[threadIdx.x] = 0.0f;
}

// ---------------------------------------------------------------------------
// main kernel: TDM-staged hidden tiles + bf16 WMMA gating GEMM + softmax/top2
// ---------------------------------------------------------------------------
__global__ __launch_bounds__(256) void router_main(
        const float* __restrict__ hs,      // (16384, 2048) f32
        const float* __restrict__ gw,      // (64, 2048)    f32
        float* __restrict__ out,           // weights | indices | aux
        float* __restrict__ psum,          // ws[0..63]
        float* __restrict__ pcnt) {        // ws[64..127]

    __shared__ __align__(16) unsigned char smem[SMEM_BYTES];
    unsigned short* sB  = (unsigned short*)smem;                  // gate bf16
    float* sHid         = (float*)(smem + GATE_BYTES);            // 2 TDM bufs
    float* sLog         = sHid;                                   // alias after GEMM
    float* sP           = (float*)(smem + GATE_BYTES + HID_BUFS_B);
    float* sC           = sP + NEXP;

    const int tid  = threadIdx.x;
    const int wave = tid >> 5;
    const int lane = tid & 31;
    const int half = lane >> 4;
    const int mrow = lane & 15;
    const int tok_base = blockIdx.x * TOKS;
    const bool leader  = (wave == 0);
    const unsigned ldsHid0 = (unsigned)(uintptr_t)(void*)(smem + GATE_BYTES);

    if (tid < NEXP) { sP[tid] = 0.0f; sC[tid] = 0.0f; }

    // Kick off the first hidden tile DMA before converting the gate.
    if (leader) tdm_load_tile(hs, tok_base, 0, ldsHid0, sHid);

    // ---- Phase 1: gate_w (f32) -> LDS bf16, padded pitch -------------------
    {
        const float4* g4 = (const float4*)gw;              // 32768 float4s
        #pragma unroll 4
        for (int i = 0; i < 128; ++i) {
            int q = tid + 256 * i;
            float4 v = g4[q];
            int f = q << 2;
            int e = f >> 11;
            int k = f & (HID - 1);
            uint2 pk;
            pk.x = (unsigned int)f32_to_bf16_rne(v.x) |
                   ((unsigned int)f32_to_bf16_rne(v.y) << 16);
            pk.y = (unsigned int)f32_to_bf16_rne(v.z) |
                   ((unsigned int)f32_to_bf16_rne(v.w) << 16);
            *(uint2*)&sB[e * GATE_PITCH + k] = pk;
        }
    }
    if (leader) tdm_wait();            // tile 0 resident
    __syncthreads();                   // publishes gate + tile 0

    // ---- Phase 2: WMMA GEMM, double-buffered TDM pipeline ------------------
    v8f acc[4];
    #pragma unroll
    for (int nt = 0; nt < 4; ++nt)
        acc[nt] = (v8f){0.f, 0.f, 0.f, 0.f, 0.f, 0.f, 0.f, 0.f};

    const int rowBase = (wave * 16 + mrow) * HID_PITCH_DW + half * 8;

    for (int i = 0; i < HID / KCHUNK; ++i) {
        const int buf = i & 1;
        if (leader && (i + 1) < HID / KCHUNK)
            tdm_load_tile(hs, tok_base, (i + 1) * KCHUNK,
                          ldsHid0 + (unsigned)((buf ^ 1) * HID_TILE_B),
                          sHid + (buf ^ 1) * (HID_TILE_B / 4));

        // A: 16x32 bf16 fragment from the padded LDS tile.
        // lane<16: K {0..7, 16..23}; lane>=16: K {8..15, 24..31}
        const float* hrow = sHid + buf * (HID_TILE_B / 4) + rowBase;
        float4 a0 = ((const float4*)hrow)[0];
        float4 a1 = ((const float4*)hrow)[1];
        float4 a2 = ((const float4*)(hrow + 16))[0];
        float4 a3 = ((const float4*)(hrow + 16))[1];

        FragB16 A;
        A.u[0]  = f32_to_bf16_rne(a0.x); A.u[1]  = f32_to_bf16_rne(a0.y);
        A.u[2]  = f32_to_bf16_rne(a0.z); A.u[3]  = f32_to_bf16_rne(a0.w);
        A.u[4]  = f32_to_bf16_rne(a1.x); A.u[5]  = f32_to_bf16_rne(a1.y);
        A.u[6]  = f32_to_bf16_rne(a1.z); A.u[7]  = f32_to_bf16_rne(a1.w);
        A.u[8]  = f32_to_bf16_rne(a2.x); A.u[9]  = f32_to_bf16_rne(a2.y);
        A.u[10] = f32_to_bf16_rne(a2.z); A.u[11] = f32_to_bf16_rne(a2.w);
        A.u[12] = f32_to_bf16_rne(a3.x); A.u[13] = f32_to_bf16_rne(a3.y);
        A.u[14] = f32_to_bf16_rne(a3.z); A.u[15] = f32_to_bf16_rne(a3.w);

        // B: 32x16 bf16 per N-tile from gate LDS.
        const int kb = i * KCHUNK + half * 8;
        #pragma unroll
        for (int nt = 0; nt < 4; ++nt) {
            FragB16 Bf;
            const uint4* bp =
                (const uint4*)&sB[(nt * 16 + mrow) * GATE_PITCH + kb];
            Bf.q[0] = bp[0];
            Bf.q[1] = bp[2];
            acc[nt] = __builtin_amdgcn_wmma_f32_16x16x32_bf16(
                false, A.v, false, Bf.v, (short)0, acc[nt], false, false);
        }

        if (leader) tdm_wait();        // next tile resident
        __syncthreads();               // publish; also guards buf reuse
    }

    // ---- Phase 2.5: scatter D tiles to sLog (aliases hid buffers) ----------
    #pragma unroll
    for (int nt = 0; nt < 4; ++nt) {
        #pragma unroll
        for (int r = 0; r < 8; ++r) {
            int tl = wave * 16 + r + half * 8;
            int ex = nt * 16 + mrow;
            sLog[tl * LOG_PITCH + ex] = acc[nt][r];
        }
    }
    __syncthreads();

    // ---- Phase 3: per-token softmax + top-2 + partial aux ------------------
    if (tid < TOKS) {
        float* L = &sLog[tid * LOG_PITCH];
        float mx = -3.4e38f;
        for (int e = 0; e < NEXP; ++e) mx = fmaxf(mx, L[e]);

        float s = 0.f, v1 = 0.f, v2 = 0.f;
        int i1 = 0, i2 = 0;
        for (int e = 0; e < NEXP; ++e) {
            float p = __expf(L[e] - mx);
            L[e] = p;
            s += p;
            if (p > v1)      { v2 = v1; i2 = i1; v1 = p; i1 = e; }
            else if (p > v2) { v2 = p;  i2 = e; }
        }
        const float inv_s = 1.0f / s;
        for (int e = 0; e < NEXP; ++e)
            atomicAdd(&sP[e], L[e] * inv_s);
        atomicAdd(&sC[i1], 1.0f);
        atomicAdd(&sC[i2], 1.0f);

        const float p1 = v1 * inv_s, p2 = v2 * inv_s;
        const float dn = 1.0f / (p1 + p2 + 1e-9f);
        const int tok = tok_base + tid;
        out[tok * 2 + 0] = p1 * dn;
        out[tok * 2 + 1] = p2 * dn;
        out[2 * T_TOKENS + tok * 2 + 0] = (float)i1;
        out[2 * T_TOKENS + tok * 2 + 1] = (float)i2;
    }
    __syncthreads();

    if (tid < NEXP) {
        atomicAdd(&psum[tid], sP[tid]);
        atomicAdd(&pcnt[tid], sC[tid]);
    }
}

// ---------------------------------------------------------------------------
// aux loss: 0.01 * 64 * sum_e (count_e / (2T + eps)) * (psum_e / T)
// ---------------------------------------------------------------------------
__global__ void router_aux(const float* __restrict__ psum,
                           const float* __restrict__ pcnt,
                           float* __restrict__ out) {
    __shared__ float sh[NEXP];
    const int e = threadIdx.x;
    const float frac = pcnt[e] / ((float)(2 * T_TOKENS) + 1e-9f);
    const float pm   = psum[e] * (1.0f / (float)T_TOKENS);
    sh[e] = frac * pm;
    __syncthreads();
    if (e == 0) {
        float s = 0.f;
        for (int i = 0; i < NEXP; ++i) s += sh[i];
        out[4 * T_TOKENS] = 0.01f * (float)NEXP * s;
    }
}

extern "C" void kernel_launch(void* const* d_in, const int* in_sizes, int n_in,
                              void* d_out, int out_size, void* d_ws, size_t ws_size,
                              hipStream_t stream) {
    (void)in_sizes; (void)n_in; (void)out_size; (void)ws_size;
    const float* hs = (const float*)d_in[0];   // (4,4096,2048) f32
    const float* gw = (const float*)d_in[1];   // (64,2048)     f32
    float* out = (float*)d_out;
    float* ws  = (float*)d_ws;

    router_zero_ws<<<1, 128, 0, stream>>>(ws);
    router_main<<<T_TOKENS / TOKS, 256, 0, stream>>>(hs, gw, out, ws, ws + NEXP);
    router_aux<<<1, NEXP, 0, stream>>>(ws, ws + NEXP, out);
}
